// CvTSelfAttentionBlock_4011499455079
// MI455X (gfx1250) — matrix operand order repaired and checked by reading, not compiled, not run on hardware
//
#include <hip/hip_runtime.h>
#include <hip/hip_bf16.h>
#include <math.h>

// ---------------------------------------------------------------------------
// CvT self-attention block for MI455X (gfx1250, wave32, WMMA bf16 path)
// Memory-bound op (~40MB HBM floor): bf16 intermediates, L2-resident scratch,
// WMMA for all GEMMs, async-to-LDS staging for K tiles in attention.
// ---------------------------------------------------------------------------

typedef __attribute__((ext_vector_type(16))) __bf16 v16bf;
typedef __attribute__((ext_vector_type(8)))  float  v8f;

#define BATCH 8
#define LQ    3136
#define SQ    56
#define SK    28
#define LK    784
#define CCH   192
#define HEADS 3
#define HD    64
#define EPSBN 1e-5f

// ---------------------------------------------------------------------------
// Kernel 1: depthwise 3x3 conv (SAME) + inference BN, output bf16
//   stride 1: pad = 1 (56 -> 56);  stride 2: pad = 0 (56 -> 28, JAX SAME)
// ---------------------------------------------------------------------------
__global__ void dwconv_bn_kernel(const float* __restrict__ x,
                                 const float* __restrict__ dw,     // [3,3,1,C]
                                 const float* __restrict__ scale,
                                 const float* __restrict__ bias,
                                 const float* __restrict__ mean,
                                 const float* __restrict__ var,
                                 __bf16* __restrict__ out,         // [B,Ho,Ho,C]
                                 int stride, int pad, int Ho)
{
    int idx = blockIdx.x * blockDim.x + threadIdx.x;
    int total = BATCH * Ho * Ho * CCH;
    if (idx >= total) return;
    int c  = idx % CCH;
    int t  = idx / CCH;
    int xo = t % Ho; t /= Ho;
    int yo = t % Ho;
    int b  = t / Ho;

    float acc = 0.f;
#pragma unroll
    for (int ky = 0; ky < 3; ++ky) {
        int iy = yo * stride + ky - pad;
        if (iy < 0 || iy >= SQ) continue;
#pragma unroll
        for (int kx = 0; kx < 3; ++kx) {
            int ix = xo * stride + kx - pad;
            if (ix < 0 || ix >= SQ) continue;
            acc += x[((size_t)(b * SQ + iy) * SQ + ix) * CCH + c] * dw[(ky * 3 + kx) * CCH + c];
        }
    }
    float r = (acc - mean[c]) * rsqrtf(var[c] + EPSBN) * scale[c] + bias[c];
    out[idx] = (__bf16)r;
}

// ---------------------------------------------------------------------------
// Kernel 2: pack fp32 weight [192,192] (KxN) into bf16 WMMA B-fragment order:
//   Wp[((ntile*6 + kchunk)*32 + lane)*16 + e]  -> one contiguous 32B run/lane
//   lane: h = lane>>4, n = lane&15;  element pairs are K rows (2g+16h, +1)
// ---------------------------------------------------------------------------
__global__ void pack_w_kernel(const float* __restrict__ W, __bf16* __restrict__ Wp)
{
    int idx = blockIdx.x * blockDim.x + threadIdx.x;    // 12*6*32 = 2304
    if (idx >= 12 * 6 * 32) return;
    int lane = idx & 31;
    int c    = (idx >> 5) % 6;
    int t    = idx / (32 * 6);
    int h = lane >> 4, n = lane & 15;
    __bf16* o = Wp + (size_t)idx * 16;
#pragma unroll
    for (int g = 0; g < 8; ++g) {
        int kr = 32 * c + 2 * g + 16 * h;
        o[2 * g]     = (__bf16)W[(size_t)kr * CCH + 16 * t + n];
        o[2 * g + 1] = (__bf16)W[(size_t)(kr + 1) * CCH + 16 * t + n];
    }
}

// ---------------------------------------------------------------------------
// Kernel 3: pointwise GEMM  Y[M,192] = H[M,192] x W[192,192], WMMA bf16.
// One wave per 16x16 tile, K=192 in 6 WMMA steps; B from packed Wp (b128s).
// transposed!=0: scatter-store Y as Yt[(b*CCH + col)*LK + kv]   (for V)
// ---------------------------------------------------------------------------
__global__ void pw_gemm_kernel(const __bf16* __restrict__ A,   // [M,192]
                               const __bf16* __restrict__ Wp,  // packed fragments
                               __bf16* __restrict__ Y,
                               float oscale, int transposed)
{
    int lane = threadIdx.x & 31;
    int h = lane >> 4;
    int r = lane & 15;
    int m0 = blockIdx.x * 16;
    int n0 = blockIdx.y * 16;

    v8f acc = {};
    for (int k0 = 0; k0 < CCH; k0 += 32) {
        v16bf a, b;
        // A: row (m0+r), one contiguous 32B run: [k0+16h, +16)
        const __bf16* ap = A + (size_t)(m0 + r) * CCH + k0 + 16 * h;
#pragma unroll
        for (int i = 0; i < 8; ++i) { a[i] = ap[i]; a[8 + i] = ap[8 + i]; }
        // B: one contiguous 32B run per lane from the packed weight
        const __bf16* bp = Wp + (((size_t)(n0 >> 4) * 6 + (k0 >> 5)) * 32 + lane) * 16;
#pragma unroll
        for (int i = 0; i < 16; ++i) b[i] = bp[i];
        acc = __builtin_amdgcn_wmma_f32_16x16x32_bf16(false, a, false, b,
                                                      (short)0, acc, false, false);
    }
#pragma unroll
    for (int j = 0; j < 8; ++j) {
        int row = m0 + j + 8 * h;
        int col = n0 + r;
        __bf16 v = (__bf16)(acc[j] * oscale);
        if (!transposed) {
            Y[(size_t)row * CCH + col] = v;
        } else {
            int bb = row / LK, kv = row % LK;   // tile never straddles a batch (784%16==0)
            Y[((size_t)bb * CCH + col) * LK + kv] = v;
        }
    }
}

// ---------------------------------------------------------------------------
// Kernel 4: attention, online softmax, one wave per (batch, head, 16-q tile).
// K tile (16x64 bf16, 2KB) staged to LDS with global_load_async_to_lds_b128;
// V read from kv-major Vt so B-fragments are contiguous b128 loads;
// P transposed C->A layout through LDS (upper 16 K-columns pre-zeroed).
// ---------------------------------------------------------------------------
__global__ void attn_kernel(const __bf16* __restrict__ Q,   // [B,3136,192] (pre-scaled)
                            const __bf16* __restrict__ K,   // [B,784,192]
                            const __bf16* __restrict__ Vt,  // [B,192,784] kv-major (+pad)
                            float* __restrict__ O)          // [B,3136,192] f32
{
    __shared__ __bf16 Klds[16 * 64];
    __shared__ __bf16 Plds[16 * 32];

    int lane = threadIdx.x & 31;
    int h = lane >> 4;
    int r = lane & 15;
    int q0   = blockIdx.x * 16;  // 196 tiles
    int head = blockIdx.y;       // 3
    int b    = blockIdx.z;       // 8

    const __bf16* qbase = Q + ((size_t)b * LQ + q0) * CCH + head * HD;
    const __bf16* kbase = K + (size_t)b * LK * CCH + head * HD;
    const __bf16* vtb   = Vt + ((size_t)b * CCH + head * HD) * LK;

    unsigned klds0 = (unsigned)(uintptr_t)&Klds[0];

    // zero P staging once; cols 16..31 stay zero -> zero-padded A fragment
    for (int i = threadIdx.x; i < 16 * 32; i += 32) Plds[i] = (__bf16)0.f;
    __syncthreads();

    // Q A-fragments for K-dim [0,32) and [32,64): one 32B run each
    v16bf aq0, aq1;
    {
        const __bf16* qp = qbase + (size_t)r * CCH + 16 * h;
#pragma unroll
        for (int i = 0; i < 8; ++i) {
            aq0[i]     = qp[i];
            aq0[8 + i] = qp[8 + i];
            aq1[i]     = qp[32 + i];
            aq1[8 + i] = qp[40 + i];
        }
    }

    float m[8], l[8];
    v8f acc[4];
#pragma unroll
    for (int j = 0; j < 8; ++j) { m[j] = -1e30f; l[j] = 0.f; }
    {
        v8f z = {};
#pragma unroll
        for (int nt = 0; nt < 4; ++nt) acc[nt] = z;
    }

    for (int kv0 = 0; kv0 < LK; kv0 += 16) {
        // ---- async-copy K tile rows [kv0, kv0+16) x 64ch into LDS ----
        // per issue: 32 lanes x 16B = 4 rows; 4 issues = 16 rows (2KB)
        {
            int row = lane >> 3;       // 0..3
            int seg = lane & 7;        // 8 x 16B = 128B per row
#pragma unroll
            for (int i = 0; i < 4; ++i) {
                const __bf16* g = kbase + (size_t)(kv0 + i * 4 + row) * CCH + seg * 8;
                unsigned long long ga = (unsigned long long)(uintptr_t)g;
                unsigned la = klds0 + (unsigned)(((i * 4 + row) * 64 + seg * 8) * 2);
                asm volatile("global_load_async_to_lds_b128 %0, %1, off"
                             :: "v"(la), "v"(ga) : "memory");
            }
            asm volatile("s_wait_asynccnt 0x0" ::: "memory");
        }
        __syncthreads();

        // ---- scores: S = Q x K^T ; B[d][n] = Klds[n][d], contiguous runs ----
        v16bf bk0, bk1;
        {
            const __bf16* kl = &Klds[r * 64 + 16 * h];
#pragma unroll
            for (int i = 0; i < 16; ++i) {
                bk0[i] = kl[i];
                bk1[i] = kl[32 + i];
            }
        }
        v8f s = {};
        s = __builtin_amdgcn_wmma_f32_16x16x32_bf16(false, aq0, false, bk0, (short)0, s, false, false);
        s = __builtin_amdgcn_wmma_f32_16x16x32_bf16(false, aq1, false, bk1, (short)0, s, false, false);

        // ---- online softmax (row j+8h lives across the 16 lanes of a half) ----
        float p[8];
#pragma unroll
        for (int j = 0; j < 8; ++j) {
            float sv = s[j];
            float rowmax = sv;
            rowmax = fmaxf(rowmax, __shfl_xor(rowmax, 1, 32));
            rowmax = fmaxf(rowmax, __shfl_xor(rowmax, 2, 32));
            rowmax = fmaxf(rowmax, __shfl_xor(rowmax, 4, 32));
            rowmax = fmaxf(rowmax, __shfl_xor(rowmax, 8, 32));
            float mnew = fmaxf(m[j], rowmax);
            float corr = __expf(m[j] - mnew);
            float pe   = __expf(sv - mnew);
            float rowsum = pe;
            rowsum += __shfl_xor(rowsum, 1, 32);
            rowsum += __shfl_xor(rowsum, 2, 32);
            rowsum += __shfl_xor(rowsum, 4, 32);
            rowsum += __shfl_xor(rowsum, 8, 32);
            l[j] = l[j] * corr + rowsum;
            m[j] = mnew;
            p[j] = pe;
#pragma unroll
            for (int nt = 0; nt < 4; ++nt) acc[nt][j] *= corr;
        }

        // ---- transpose P (C layout -> A layout) through LDS ----
        __syncthreads();   // previous iteration's A-fragment reads complete
#pragma unroll
        for (int j = 0; j < 8; ++j) {
            Plds[(j + 8 * h) * 32 + r] = (__bf16)p[j];
        }
        __syncthreads();

        v16bf apf;
        {
            const __bf16* pl = &Plds[r * 32 + 16 * h];
#pragma unroll
            for (int i = 0; i < 8; ++i) { apf[i] = pl[i]; apf[8 + i] = pl[8 + i]; }
        }

        // ---- O += P x V : Vt is kv-major -> contiguous 32B fragment runs ----
        // kvr>=16 multiplies zero A columns (reads land in padded allocation).
#pragma unroll
        for (int nt = 0; nt < 4; ++nt) {
            v16bf bv;
            const __bf16* vp = vtb + (size_t)(nt * 16 + r) * LK + kv0 + 16 * h;
#pragma unroll
            for (int i = 0; i < 16; ++i) bv[i] = vp[i];
            acc[nt] = __builtin_amdgcn_wmma_f32_16x16x32_bf16(false, apf, false, bv,
                                                              (short)0, acc[nt], false, false);
        }
    }

    // ---- normalize and store fp32 output [B,3136,192] ----
    float* obase = O + ((size_t)b * LQ + q0) * CCH + head * HD;
#pragma unroll
    for (int j = 0; j < 8; ++j) {
        float invl = 1.0f / l[j];
        int row = j + 8 * h;
#pragma unroll
        for (int nt = 0; nt < 4; ++nt) {
            obase[(size_t)row * CCH + nt * 16 + r] = acc[nt][j] * invl;
        }
    }
}

// ---------------------------------------------------------------------------
// Host-side launch
// ---------------------------------------------------------------------------
static inline size_t align256(size_t v) { return (v + 255) & ~(size_t)255; }

extern "C" void kernel_launch(void* const* d_in, const int* in_sizes, int n_in,
                              void* d_out, int out_size, void* d_ws, size_t ws_size,
                              hipStream_t stream)
{
    (void)in_sizes; (void)n_in; (void)out_size; (void)ws_size;

    const float* x = (const float*)d_in[0];
    const float* q_dw = (const float*)d_in[1];
    const float* q_sc = (const float*)d_in[2];
    const float* q_bi = (const float*)d_in[3];
    const float* q_mn = (const float*)d_in[4];
    const float* q_vr = (const float*)d_in[5];
    const float* q_pw = (const float*)d_in[6];
    const float* k_dw = (const float*)d_in[7];
    const float* k_sc = (const float*)d_in[8];
    const float* k_bi = (const float*)d_in[9];
    const float* k_mn = (const float*)d_in[10];
    const float* k_vr = (const float*)d_in[11];
    const float* k_pw = (const float*)d_in[12];
    const float* v_dw = (const float*)d_in[13];
    const float* v_sc = (const float*)d_in[14];
    const float* v_bi = (const float*)d_in[15];
    const float* v_mn = (const float*)d_in[16];
    const float* v_vr = (const float*)d_in[17];
    const float* v_pw = (const float*)d_in[18];
    float* out = (float*)d_out;

    // workspace layout (bf16 intermediates; everything stays L2-resident)
    const size_t nW  = (size_t)CCH * CCH;            // 36864
    const size_t nHq = (size_t)BATCH * LQ * CCH;     // 4,816,896
    const size_t nHk = (size_t)BATCH * LK * CCH;     // 1,204,224
    char* ws = (char*)d_ws;
    size_t off = 0;
    __bf16* Wq = (__bf16*)(ws + off); off = align256(off + nW * 2);
    __bf16* Wk = (__bf16*)(ws + off); off = align256(off + nW * 2);
    __bf16* Wv = (__bf16*)(ws + off); off = align256(off + nW * 2);
    __bf16* Hq = (__bf16*)(ws + off); off = align256(off + nHq * 2);
    __bf16* Hk = (__bf16*)(ws + off); off = align256(off + nHk * 2);
    __bf16* Hv = (__bf16*)(ws + off); off = align256(off + nHk * 2);
    __bf16* Qb = (__bf16*)(ws + off); off = align256(off + nHq * 2);
    __bf16* Kb = (__bf16*)(ws + off); off = align256(off + nHk * 2);
    __bf16* Vt = (__bf16*)(ws + off); off = align256(off + nHk * 2 + 256); // +pad

    // 1) pack pointwise weights into B-fragment order (bf16)
    {
        int n = 12 * 6 * 32;
        pack_w_kernel<<<(n + 255) / 256, 256, 0, stream>>>(q_pw, Wq);
        pack_w_kernel<<<(n + 255) / 256, 256, 0, stream>>>(k_pw, Wk);
        pack_w_kernel<<<(n + 255) / 256, 256, 0, stream>>>(v_pw, Wv);
    }

    // 2) depthwise conv + BN
    {
        int totQ = (int)nHq;
        dwconv_bn_kernel<<<(totQ + 255) / 256, 256, 0, stream>>>(
            x, q_dw, q_sc, q_bi, q_mn, q_vr, Hq, /*stride*/1, /*pad*/1, /*Ho*/SQ);
        int totK = (int)nHk;
        dwconv_bn_kernel<<<(totK + 255) / 256, 256, 0, stream>>>(
            x, k_dw, k_sc, k_bi, k_mn, k_vr, Hk, 2, 0, SK);
        dwconv_bn_kernel<<<(totK + 255) / 256, 256, 0, stream>>>(
            x, v_dw, v_sc, v_bi, v_mn, v_vr, Hv, 2, 0, SK);
    }

    // 3) pointwise GEMMs via WMMA (q folded with 1/sqrt(64); V stored kv-major)
    {
        dim3 gq(BATCH * LQ / 16, CCH / 16);
        pw_gemm_kernel<<<gq, 32, 0, stream>>>(Hq, Wq, Qb, 0.125f, 0);
        dim3 gk(BATCH * LK / 16, CCH / 16);
        pw_gemm_kernel<<<gk, 32, 0, stream>>>(Hk, Wk, Kb, 1.0f, 0);
        pw_gemm_kernel<<<gk, 32, 0, stream>>>(Hv, Wv, Vt, 1.0f, 1);
    }

    // 4) attention (online softmax, WMMA, async K staging)
    {
        dim3 ga(LQ / 16, HEADS, BATCH);
        attn_kernel<<<ga, 32, 0, stream>>>(Qb, Kb, Vt, out);
    }
}
